// DeprecatedMixtureOfExperts_37606733644550
// MI455X (gfx1250) — matrix operand order repaired and checked by reading, not compiled, not run on hardware
//
#include <hip/hip_runtime.h>
#include <stdint.h>

#define D_IN   256
#define D_HID  512
#define D_OUT  256
#define N_EXP  64
#define T_TOT  (8 * 4096)

#define TILE_T    64
#define XS_STRIDE (D_IN + 8)    // 264 bf16, row shift = 16B -> conflict-free fragment reads
#define HS_STRIDE (D_HID + 8)   // 520 bf16

typedef __attribute__((ext_vector_type(16))) __bf16 bf16x16;
typedef __attribute__((ext_vector_type(8)))  float  f32x8;
typedef int v4i __attribute__((vector_size(16)));   // matches async-LDS builtin param type

#define AS1 __attribute__((address_space(1)))
#define AS3 __attribute__((address_space(3)))

union BFrag { uint4 u[2]; bf16x16 v; };

// gfx1250 async global->LDS copy (ASYNCcnt-tracked), guarded so compile never breaks
#if defined(__has_builtin)
#if __has_builtin(__builtin_amdgcn_global_load_async_to_lds_b128)
#define ASYNC_STAGE 1
#endif
#endif

__device__ __forceinline__ void wait_asynccnt0() {
#if defined(__has_builtin)
#if __has_builtin(__builtin_amdgcn_s_wait_asynccnt)
  __builtin_amdgcn_s_wait_asynccnt(0);
  return;
#else
  asm volatile("s_wait_asynccnt 0x0" ::: "memory");
  return;
#endif
#else
  asm volatile("s_wait_asynccnt 0x0" ::: "memory");
#endif
}

__device__ __forceinline__ unsigned short f2bf(float f) {
  union { float f; unsigned u; } c; c.f = f;
  unsigned u = c.u;
  unsigned r = u + 0x7FFFu + ((u >> 16) & 1u);  // round-to-nearest-even
  return (unsigned short)(r >> 16);
}

// -------------------- init kernels --------------------
__global__ void k_zero_out(float* __restrict__ out, int n4) {
  int i = blockIdx.x * 256 + threadIdx.x;
  if (i < n4) ((float4*)out)[i] = make_float4(0.f, 0.f, 0.f, 0.f);
}

__global__ void k_zero_meta(int* __restrict__ counts) {
  if (threadIdx.x < N_EXP) counts[threadIdx.x] = 0;
}

// -------------------- router: logits, top-2, softmax gates --------------------
__global__ void __launch_bounds__(32) k_router(const float* __restrict__ x,
                                               const float* __restrict__ Wr,
                                               const float* __restrict__ br,
                                               int2*  __restrict__ idx2,
                                               float2* __restrict__ gate2,
                                               int* __restrict__ counts) {
  __shared__ float xs[32][D_IN + 1];
  const int tid   = threadIdx.x;
  const int tbase = blockIdx.x * 32;

  for (int i = tid; i < 32 * (D_IN / 4); i += 32) {
    int row = i >> 6;            // D_IN/4 = 64 chunks per row
    int c4  = i & 63;
    float4 v = *(const float4*)(x + (size_t)(tbase + row) * D_IN + c4 * 4);
    xs[row][c4 * 4 + 0] = v.x; xs[row][c4 * 4 + 1] = v.y;
    xs[row][c4 * 4 + 2] = v.z; xs[row][c4 * 4 + 3] = v.w;
  }
  __syncthreads();

  float best0 = -3.4e38f, best1 = -3.4e38f;
  int   i0 = 0, i1 = 0;
  for (int e = 0; e < N_EXP; ++e) {
    float acc = br[e];                       // uniform (broadcast) load
    const float* wc = Wr + e;                // Wr[d][e], stride N_EXP; uniform per step
    #pragma unroll 8
    for (int d = 0; d < D_IN; ++d)
      acc += xs[tid][d] * wc[(size_t)d * N_EXP];
    if (acc > best0)      { best1 = best0; i1 = i0; best0 = acc; i0 = e; }
    else if (acc > best1) { best1 = acc; i1 = e; }
  }
  float ed = __expf(best1 - best0);          // best1 <= best0: stable
  float g0 = 1.f / (1.f + ed);
  float g1 = 1.f - g0;

  int t = tbase + tid;
  idx2[t]  = make_int2(i0, i1);
  gate2[t] = make_float2(g0, g1);
  atomicAdd(&counts[i0], 1);
  atomicAdd(&counts[i1], 1);
}

// -------------------- scan: offsets + tile table --------------------
__global__ void k_scan(const int* __restrict__ counts,
                       int* __restrict__ tok_off, int* __restrict__ tile_off,
                       int* __restrict__ cursors, int* __restrict__ total) {
  if (blockIdx.x == 0 && threadIdx.x == 0) {
    int s = 0, ts = 0;
    for (int e = 0; e < N_EXP; ++e) {
      tok_off[e]  = s;  s  += counts[e];
      tile_off[e] = ts; ts += (counts[e] + TILE_T - 1) / TILE_T;
      cursors[e]  = 0;
    }
    tok_off[N_EXP]  = s;
    tile_off[N_EXP] = ts;
    *total = ts;
  }
}

// -------------------- scatter: build per-expert token lists --------------------
__global__ void k_scatter(const int2* __restrict__ idx2, const float2* __restrict__ gate2,
                          const int* __restrict__ tok_off, int* __restrict__ cursors,
                          int* __restrict__ tok_list, float* __restrict__ gate_list) {
  int t = blockIdx.x * 256 + threadIdx.x;
  if (t >= T_TOT) return;
  int2   id = idx2[t];
  float2 g  = gate2[t];
  int p0 = atomicAdd(&cursors[id.x], 1);
  tok_list[tok_off[id.x] + p0]  = t;
  gate_list[tok_off[id.x] + p0] = g.x;
  int p1 = atomicAdd(&cursors[id.y], 1);
  tok_list[tok_off[id.y] + p1]  = t;
  gate_list[tok_off[id.y] + p1] = g.y;
}

// -------------------- fp32 -> bf16 convert (x) --------------------
__global__ void k_cvt_x(const float* __restrict__ x, unsigned short* __restrict__ xb, int n) {
  int i = (blockIdx.x * 256 + threadIdx.x) * 4;
  if (i >= n) return;
  float4 v = *(const float4*)(x + i);
  union { unsigned short s[4]; uint2 u; } o;
  o.s[0] = f2bf(v.x); o.s[1] = f2bf(v.y); o.s[2] = f2bf(v.z); o.s[3] = f2bf(v.w);
  *(uint2*)(xb + i) = o.u;
}

// transpose + convert: src [E][K][N] f32 -> dst [E][N][K] bf16
__global__ void k_transcvt(const float* __restrict__ src, unsigned short* __restrict__ dst,
                           int K, int N) {
  __shared__ float tile[32][33];
  int e  = blockIdx.z;
  int n0 = blockIdx.x * 32, k0 = blockIdx.y * 32;
  int tx = threadIdx.x, ty = threadIdx.y;
  const float*    s = src + (size_t)e * K * N;
  unsigned short* d = dst + (size_t)e * N * K;
  #pragma unroll
  for (int j = 0; j < 4; ++j)
    tile[ty + j * 8][tx] = s[(size_t)(k0 + ty + j * 8) * N + n0 + tx];
  __syncthreads();
  #pragma unroll
  for (int j = 0; j < 4; ++j)
    d[(size_t)(n0 + ty + j * 8) * K + k0 + tx] = f2bf(tile[tx][ty + j * 8]);
}

// -------------------- fused expert FFN (WMMA bf16) --------------------
__global__ void __launch_bounds__(256) k_expert_ffn(
    const unsigned short* __restrict__ xb,    // [T][256] bf16
    const unsigned short* __restrict__ w1t,   // [E][512][256] bf16 (N-major)
    const unsigned short* __restrict__ w2t,   // [E][256][512] bf16 (N-major)
    const float* __restrict__ b1,             // [E][512]
    const float* __restrict__ b2,             // [E][256]
    const int*  __restrict__ tok_off,         // [E+1]
    const int*  __restrict__ tile_off,        // [E+1]
    const int*  __restrict__ total_tiles,
    const int*  __restrict__ tok_list,
    const float* __restrict__ gate_list,
    float* __restrict__ out) {

  __shared__ unsigned short xs[TILE_T * XS_STRIDE];
  __shared__ unsigned short hs[TILE_T * HS_STRIDE];
  __shared__ int   s_ts[TILE_T];
  __shared__ float s_gs[TILE_T];
  __shared__ int   s_meta[4];

  const int tid  = threadIdx.x;
  const int lane = tid & 31;
  const int wv   = tid >> 5;
  const int lrow = lane & 15;
  const int lhi  = lane >> 4;
  const int ntiles = *total_tiles;

  for (int tile = (int)blockIdx.x; tile < ntiles; tile += (int)gridDim.x) {
    if (tid == 0) {
      int e = 0;
      while (tile >= tile_off[e + 1]) ++e;          // find owning expert
      int lt    = tile - tile_off[e];
      int start = tok_off[e] + lt * TILE_T;
      int nv    = tok_off[e + 1] - start; if (nv > TILE_T) nv = TILE_T;
      s_meta[0] = e; s_meta[1] = start; s_meta[2] = nv;
    }
    __syncthreads();
    const int e = s_meta[0], start = s_meta[1], nv = s_meta[2];

    if (tid < TILE_T) {
      if (tid < nv) { s_ts[tid] = tok_list[start + tid]; s_gs[tid] = gate_list[start + tid]; }
      else          { s_ts[tid] = 0; s_gs[tid] = 0.f; }
    }
    // stage gathered x rows as bf16 (zero-fill invalid rows)
    for (int i = tid; i < TILE_T * (D_IN / 8); i += 256) {
      int row = i >> 5;            // 32 chunks of 8 bf16 per row
      int c   = i & 31;
#ifdef ASYNC_STAGE
      if (row < nv) {
        int t = tok_list[start + row];
        __builtin_amdgcn_global_load_async_to_lds_b128(
            (AS1 v4i*)(xb + (size_t)t * D_IN + c * 8),
            (AS3 v4i*)(xs + row * XS_STRIDE + c * 8),
            0, 0);
      } else {
        *(uint4*)(xs + row * XS_STRIDE + c * 8) = make_uint4(0, 0, 0, 0);
      }
#else
      uint4 v = make_uint4(0, 0, 0, 0);
      if (row < nv) {
        int t = tok_list[start + row];
        v = *(const uint4*)(xb + (size_t)t * D_IN + c * 8);
      }
      *(uint4*)(xs + row * XS_STRIDE + c * 8) = v;
#endif
    }
#ifdef ASYNC_STAGE
    wait_asynccnt0();
#endif
    __syncthreads();

    const unsigned short* w1e = w1t + (size_t)e * D_HID * D_IN;
    const unsigned short* w2e = w2t + (size_t)e * D_OUT * D_HID;
    const float* b1e = b1 + e * D_HID;
    const float* b2e = b2 + e * D_OUT;

    // prefetch this wave's W2 slice (32 rows x 1KB) while GEMM1 runs:
    // one instruction prefetches 32 lane addresses -> 8 instructions cover 32KB
    {
      const char* p = (const char*)(w2e + (size_t)(wv * 32 + lane) * D_HID);
      #pragma unroll
      for (int j = 0; j < 8; ++j)
        __builtin_prefetch(p + j * 128, 0, 3);
    }

    // ---- GEMM1: h = relu(x @ W1 + b1), M=64 N=512 K=256 ----
    for (int nt = 0; nt < 4; ++nt) {
      const int n0 = wv * 64 + nt * 16;
      BFrag Bk[8];
      const unsigned short* bp = w1e + (size_t)(n0 + lrow) * D_IN + lhi * 16;
      #pragma unroll
      for (int k = 0; k < 8; ++k) {                // B: 16 contiguous K per lane
        Bk[k].u[0] = *(const uint4*)(bp + k * 32);
        Bk[k].u[1] = *(const uint4*)(bp + k * 32 + 8);
      }
      const float bias = b1e[n0 + lrow];
      #pragma unroll
      for (int mt = 0; mt < 4; ++mt) {
        f32x8 acc = {};
        const unsigned short* ap = xs + (mt * 16 + lrow) * XS_STRIDE + lhi * 16;
        #pragma unroll
        for (int k = 0; k < 8; ++k) {              // A: 16 contiguous K per lane
          BFrag a;
          a.u[0] = *(const uint4*)(ap + k * 32);
          a.u[1] = *(const uint4*)(ap + k * 32 + 8);
          acc = __builtin_amdgcn_wmma_f32_16x16x32_bf16(false, a.v, false, Bk[k].v,
                                                        (short)0, acc, false, false);
        }
        const int rbase = mt * 16 + ((lane & 16) ? 8 : 0);
        #pragma unroll
        for (int r = 0; r < 8; ++r) {
          float v = acc[r] + bias;
          v = v > 0.f ? v : 0.f;
          hs[(rbase + r) * HS_STRIDE + n0 + lrow] = f2bf(v);
        }
      }
    }
    __syncthreads();

    // ---- GEMM2: y = h @ W2 + b2, M=64 N=256 K=512; out += gate*y ----
    #pragma unroll
    for (int nt = 0; nt < 2; ++nt) {
      const int n0 = wv * 32 + nt * 16;
      f32x8 acc[4] = {{}, {}, {}, {}};
      for (int kh = 0; kh < 2; ++kh) {
        BFrag Bk[8];
        const unsigned short* bp = w2e + (size_t)(n0 + lrow) * D_HID + kh * 256 + lhi * 16;
        #pragma unroll
        for (int k = 0; k < 8; ++k) {
          Bk[k].u[0] = *(const uint4*)(bp + k * 32);
          Bk[k].u[1] = *(const uint4*)(bp + k * 32 + 8);
        }
        #pragma unroll
        for (int mt = 0; mt < 4; ++mt) {
          const unsigned short* ap = hs + (mt * 16 + lrow) * HS_STRIDE + kh * 256 + lhi * 16;
          #pragma unroll
          for (int k = 0; k < 8; ++k) {
            BFrag a;
            a.u[0] = *(const uint4*)(ap + k * 32);
            a.u[1] = *(const uint4*)(ap + k * 32 + 8);
            acc[mt] = __builtin_amdgcn_wmma_f32_16x16x32_bf16(false, a.v, false, Bk[k].v,
                                                              (short)0, acc[mt], false, false);
          }
        }
      }
      const float b2v = b2e[n0 + lrow];
      #pragma unroll
      for (int mt = 0; mt < 4; ++mt) {
        const int rbase = mt * 16 + ((lane & 16) ? 8 : 0);
        #pragma unroll
        for (int r = 0; r < 8; ++r) {
          int row = rbase + r;
          if (row < nv) {
            float y = acc[mt][r] + b2v;
            atomicAdd(out + (size_t)s_ts[row] * D_OUT + n0 + lrow, s_gs[row] * y);
          }
        }
      }
    }
    __syncthreads();
  }
}

// -------------------- launcher --------------------
extern "C" void kernel_launch(void* const* d_in, const int* in_sizes, int n_in,
                              void* d_out, int out_size, void* d_ws, size_t ws_size,
                              hipStream_t stream) {
  (void)in_sizes; (void)n_in; (void)out_size; (void)ws_size;
  const float* x  = (const float*)d_in[0];
  const float* Wr = (const float*)d_in[1];
  const float* br = (const float*)d_in[2];
  const float* W1 = (const float*)d_in[3];
  const float* b1 = (const float*)d_in[4];
  const float* W2 = (const float*)d_in[5];
  const float* b2 = (const float*)d_in[6];
  float* out = (float*)d_out;

  char* ws = (char*)d_ws;
  size_t off = 0;
  auto alloc = [&](size_t bytes) -> void* {
    void* p = ws + off;
    off += (bytes + 255) & ~(size_t)255;
    return p;
  };

  unsigned short* xb   = (unsigned short*)alloc((size_t)T_TOT * D_IN * 2);          // 16 MB
  unsigned short* w1t  = (unsigned short*)alloc((size_t)N_EXP * D_HID * D_IN * 2);  // 16 MB
  unsigned short* w2t  = (unsigned short*)alloc((size_t)N_EXP * D_OUT * D_HID * 2); // 16 MB
  int2*   idx2      = (int2*)  alloc((size_t)T_TOT * sizeof(int2));
  float2* gate2     = (float2*)alloc((size_t)T_TOT * sizeof(float2));
  int*    tok_list  = (int*)   alloc((size_t)2 * T_TOT * 4);
  float*  gate_list = (float*) alloc((size_t)2 * T_TOT * 4);
  int*    counts    = (int*)   alloc(N_EXP * 4);
  int*    tok_off   = (int*)   alloc((N_EXP + 1) * 4);
  int*    tile_off  = (int*)   alloc((N_EXP + 1) * 4);
  int*    cursors   = (int*)   alloc(N_EXP * 4);
  int*    total     = (int*)   alloc(4);

  k_zero_out <<<(T_TOT * D_OUT / 4 + 255) / 256, 256, 0, stream>>>(out, T_TOT * D_OUT / 4);
  k_zero_meta<<<1, 64, 0, stream>>>(counts);
  k_router   <<<T_TOT / 32, 32, 0, stream>>>(x, Wr, br, idx2, gate2, counts);
  k_scan     <<<1, 1, 0, stream>>>(counts, tok_off, tile_off, cursors, total);
  k_scatter  <<<(T_TOT + 255) / 256, 256, 0, stream>>>(idx2, gate2, tok_off, cursors,
                                                       tok_list, gate_list);
  k_cvt_x    <<<(T_TOT * D_IN / 4 + 255) / 256, 256, 0, stream>>>(x, xb, T_TOT * D_IN);
  dim3 tb(32, 8);
  k_transcvt <<<dim3(D_HID / 32, D_IN / 32, N_EXP), tb, 0, stream>>>(W1, w1t, D_IN, D_HID);
  k_transcvt <<<dim3(D_OUT / 32, D_HID / 32, N_EXP), tb, 0, stream>>>(W2, w2t, D_HID, D_OUT);
  k_expert_ffn<<<1024, 256, 0, stream>>>(xb, w1t, w2t, b1, b2, tok_off, tile_off, total,
                                         tok_list, gate_list, out);
}
